// QLinear_22849226014947
// MI455X (gfx1250) — 8-bit matrix operand order repaired and checked by reading, compiled, not run on hardware
//
#include <hip/hip_runtime.h>

typedef __attribute__((ext_vector_type(8)))  int  v8i;
typedef __attribute__((ext_vector_type(4)))  int  v4i;
typedef __attribute__((ext_vector_type(2)))  int  v2i;
typedef __attribute__((ext_vector_type(16))) char c16;

#define NROWS 65536
#define KDIM  1024
#define NOUT  1024

__global__ void init_ws_kernel(unsigned* wsu) {
    wsu[0] = 0u;
    wsu[1] = 0u;
}

__global__ __launch_bounds__(256)
void absmax_kernel(const float* __restrict__ src, long n4, unsigned* __restrict__ dst) {
    long tid    = (long)blockIdx.x * blockDim.x + threadIdx.x;
    long stride = (long)gridDim.x * blockDim.x;
    unsigned m = 0u;
    const float4* s4 = (const float4*)src;
    for (long i = tid; i < n4; i += stride) {
        float4 v = s4[i];
        unsigned b0 = __float_as_uint(fabsf(v.x));
        unsigned b1 = __float_as_uint(fabsf(v.y));
        unsigned b2 = __float_as_uint(fabsf(v.z));
        unsigned b3 = __float_as_uint(fabsf(v.w));
        unsigned a = b0 > b1 ? b0 : b1;
        unsigned b = b2 > b3 ? b2 : b3;
        unsigned c = a > b ? a : b;
        m = m > c ? m : c;
    }
    __shared__ unsigned red[256];
    red[threadIdx.x] = m;
    __syncthreads();
    for (int off = 128; off > 0; off >>= 1) {
        if ((int)threadIdx.x < off) {
            unsigned o = red[threadIdx.x + off];
            if (o > red[threadIdx.x]) red[threadIdx.x] = o;
        }
        __syncthreads();
    }
    if (threadIdx.x == 0) atomicMax(dst, red[0]);
}

__global__ void scale_kernel(unsigned* wsu) {
    float* wsf = (float*)wsu;
    float mx = __uint_as_float(wsu[0]);
    float mw = __uint_as_float(wsu[1]);
    float ex = (mx > 0.0f) ? ceilf(log2f(mx)) : 0.0f;
    float ew = (mw > 0.0f) ? ceilf(log2f(mw)) : 0.0f;
    float sx = exp2f(7.0f - ex);
    float sw = exp2f(7.0f - ew);
    wsf[2] = 1.0f / (sx * sw);
    wsf[3] = sx;
    wsf[4] = sw;
}

__global__ __launch_bounds__(256)
void quant_kernel(const float* __restrict__ src, signed char* __restrict__ dst,
                  const float* __restrict__ scale_ptr) {
    const float s = *scale_ptr;
    long base = ((long)blockIdx.x * blockDim.x + threadIdx.x) * 16;
    const float4* s4 = (const float4*)(src + base);
    c16 out;
#pragma unroll
    for (int q = 0; q < 4; ++q) {
        float4 v = s4[q];
        float f[4] = {v.x, v.y, v.z, v.w};
#pragma unroll
        for (int j = 0; j < 4; ++j) {
            float t = rintf(f[j] * s);
            t = fminf(127.0f, fmaxf(-127.0f, t));
            out[q * 4 + j] = (signed char)(int)t;
        }
    }
    *(c16*)(dst + base) = out;
}

__global__ __launch_bounds__(256)
void gemm_iu8_kernel(const signed char* __restrict__ Aq,
                     const signed char* __restrict__ Wq,
                     float* __restrict__ C,
                     const float* __restrict__ wsf) {
    const int lane   = threadIdx.x & 31;
    const int wave   = threadIdx.x >> 5;
    const int laneLo = lane & 15;
    const int laneHi = lane >> 4;

    const int m_base = blockIdx.y * 128 + (wave >> 2) * 64;
    const int n_base = blockIdx.x * 256 + (wave & 3) * 64;

    const float inv = wsf[2];

    v8i acc[4][4];
#pragma unroll
    for (int t = 0; t < 4; ++t)
#pragma unroll
        for (int s = 0; s < 4; ++s)
#pragma unroll
            for (int v = 0; v < 8; ++v) acc[t][s][v] = 0;

    const signed char* arow[4];
#pragma unroll
    for (int t = 0; t < 4; ++t)
        arow[t] = Aq + (long)(m_base + t * 16 + laneLo) * KDIM + laneHi * 16;
    const signed char* brow[4];
#pragma unroll
    for (int s = 0; s < 4; ++s)
        brow[s] = Wq + (long)(n_base + s * 16 + laneLo) * KDIM + laneHi * 16;

    for (int k0 = 0; k0 < KDIM; k0 += 64) {
        v8i a[4];
#pragma unroll
        for (int t = 0; t < 4; ++t) {
            v2i p0 = *(const v2i*)(arow[t] + k0 + 0);
            v2i p1 = *(const v2i*)(arow[t] + k0 + 8);
            v2i p2 = *(const v2i*)(arow[t] + k0 + 32);
            v2i p3 = *(const v2i*)(arow[t] + k0 + 40);
            a[t][0] = p0.x; a[t][1] = p0.y;
            a[t][2] = p1.x; a[t][3] = p1.y;
            a[t][4] = p2.x; a[t][5] = p2.y;
            a[t][6] = p3.x; a[t][7] = p3.y;
        }
        v8i b[4];
#pragma unroll
        for (int s = 0; s < 4; ++s) {
            v4i q0 = *(const v4i*)(brow[s] + k0 + 0);
            v4i q1 = *(const v4i*)(brow[s] + k0 + 32);
            b[s][0] = q0.x; b[s][1] = q0.y; b[s][2] = q0.z; b[s][3] = q0.w;
            b[s][4] = q1.x; b[s][5] = q1.y; b[s][6] = q1.z; b[s][7] = q1.w;
        }
#pragma unroll
        for (int t = 0; t < 4; ++t)
#pragma unroll
            for (int s = 0; s < 4; ++s)
                acc[t][s] = __builtin_amdgcn_wmma_i32_16x16x64_iu8(
                     true, a[t],  true, b[s],
                    acc[t][s],  false,  false);
    }

#pragma unroll
    for (int t = 0; t < 4; ++t) {
#pragma unroll
        for (int s = 0; s < 4; ++s) {
            const int n = n_base + s * 16 + laneLo;
#pragma unroll
            for (int v = 0; v < 8; ++v) {
                const int m = m_base + t * 16 + laneHi * 8 + v;
                __builtin_nontemporal_store((float)acc[t][s][v] * inv,
                                            C + (long)m * NOUT + n);
            }
        }
    }
}

extern "C" void kernel_launch(void* const* d_in, const int* in_sizes, int n_in,
                              void* d_out, int out_size, void* d_ws, size_t ws_size,
                              hipStream_t stream) {
    const float* x = (const float*)d_in[0];
    const float* w = (const float*)d_in[1];

    const long nx = (long)in_sizes[0];
    const long nw = (long)in_sizes[1];

    unsigned*    wsu = (unsigned*)d_ws;
    const float* wsf = (const float*)d_ws;
    signed char* xq  = (signed char*)d_ws + 256;
    signed char* wq  = xq + nx;

    init_ws_kernel<<<1, 1, 0, stream>>>(wsu);

    absmax_kernel<<<2048, 256, 0, stream>>>(x, nx >> 2, wsu + 0);
    absmax_kernel<<<256,  256, 0, stream>>>(w, nw >> 2, wsu + 1);

    scale_kernel<<<1, 1, 0, stream>>>(wsu);

    quant_kernel<<<(int)(nx / (16 * 256)), 256, 0, stream>>>(x, xq, wsf + 3);
    quant_kernel<<<(int)(nw / (16 * 256)), 256, 0, stream>>>(w, wq, wsf + 4);

    dim3 grid(NOUT / 256, NROWS / 128);
    gemm_iu8_kernel<<<grid, 256, 0, stream>>>(xq, wq, (float*)d_out, wsf);
}
